// Attention_33775622816029
// MI455X (gfx1250) — matrix operand order repaired and checked by reading, not compiled, not run on hardware
//
#include <hip/hip_runtime.h>
#include <math.h>

// ---- types ----
typedef __attribute__((ext_vector_type(16))) __bf16 bf16x16;
typedef __attribute__((ext_vector_type(8)))  __bf16 bf16x8;
typedef __attribute__((ext_vector_type(8)))  float  f32x8;
typedef __attribute__((ext_vector_type(4)))  float  f32x4;

__device__ __forceinline__ __bf16 f2bf(float f) {
    unsigned u = __builtin_bit_cast(unsigned, f);
    unsigned r = (u + 0x7FFFu + ((u >> 16) & 1u)) >> 16;   // RNE
    unsigned short s = (unsigned short)r;
    return __builtin_bit_cast(__bf16, s);
}

#define CAT16(a, b) __builtin_shufflevector(a, b, 0,1,2,3,4,5,6,7,8,9,10,11,12,13,14,15)

// 32-bit LDS byte offset from a generic pointer to a __shared__ object
typedef __attribute__((address_space(3))) char lds_char;
__device__ __forceinline__ unsigned lds_off(const void* p) {
    return (unsigned)(unsigned long long)(lds_char*)p;
}

// One wave-wide async 16B/lane copy: LDS[ldsaddr] <= *(saddr + voff)
__device__ __forceinline__ void async_b128(unsigned ldsaddr, const char* sbase, unsigned voff) {
    asm volatile("global_load_async_to_lds_b128 %0, %1, %2"
                 :: "v"(ldsaddr), "v"(voff), "s"(sbase) : "memory");
}

// xor-butterfly step within 16-lane rows, pure VALU (v_permlane16_b32)
__device__ __forceinline__ float xmax16(float v, unsigned s0, unsigned s1, int xorm) {
#if __has_builtin(__builtin_amdgcn_permlane16)
    unsigned u = __builtin_bit_cast(unsigned, v);
    unsigned r = __builtin_amdgcn_permlane16(u, u, s0, s1, false, false);
    return fmaxf(v, __builtin_bit_cast(float, r));
#else
    return fmaxf(v, __shfl_xor(v, xorm, 32));
#endif
}

// =====================================================================
// Pre-pass: K(f32,[B,K,D]) -> Kb(bf16,[B,K,D]); V(f32,[B,K,D]) -> Vt(bf16,[B,D,K])
// =====================================================================
__global__ __launch_bounds__(256) void
prep_kernel(const float* __restrict__ Kp, const float* __restrict__ Vp,
            __bf16* __restrict__ Kb, __bf16* __restrict__ Vt) {
    constexpr int D = 128, NK = 2048;
    __shared__ __align__(16) __bf16 vt[128 * 32];   // [d][k]

    const int b    = blockIdx.y;
    const int key0 = blockIdx.x * 32;
    const int tid  = threadIdx.x;
    const int key_lin = tid >> 3;
    const int d_lin   = (tid & 7) * 16;

    {   // K: straight convert
        const float* ks = Kp + ((size_t)b * NK + key0 + key_lin) * D + d_lin;
        f32x4 x0 = *reinterpret_cast<const f32x4*>(ks + 0);
        f32x4 x1 = *reinterpret_cast<const f32x4*>(ks + 4);
        f32x4 x2 = *reinterpret_cast<const f32x4*>(ks + 8);
        f32x4 x3 = *reinterpret_cast<const f32x4*>(ks + 12);
        bf16x8 lo, hi;
        lo[0]=f2bf(x0[0]); lo[1]=f2bf(x0[1]); lo[2]=f2bf(x0[2]); lo[3]=f2bf(x0[3]);
        lo[4]=f2bf(x1[0]); lo[5]=f2bf(x1[1]); lo[6]=f2bf(x1[2]); lo[7]=f2bf(x1[3]);
        hi[0]=f2bf(x2[0]); hi[1]=f2bf(x2[1]); hi[2]=f2bf(x2[2]); hi[3]=f2bf(x2[3]);
        hi[4]=f2bf(x3[0]); hi[5]=f2bf(x3[1]); hi[6]=f2bf(x3[2]); hi[7]=f2bf(x3[3]);
        __bf16* kd = Kb + ((size_t)b * NK + key0 + key_lin) * D + d_lin;
        *reinterpret_cast<bf16x8*>(kd + 0) = lo;
        *reinterpret_cast<bf16x8*>(kd + 8) = hi;
    }
    {   // V: convert + transpose through LDS
        const float* vs = Vp + ((size_t)b * NK + key0 + key_lin) * D + d_lin;
        f32x4 y0 = *reinterpret_cast<const f32x4*>(vs + 0);
        f32x4 y1 = *reinterpret_cast<const f32x4*>(vs + 4);
        f32x4 y2 = *reinterpret_cast<const f32x4*>(vs + 8);
        f32x4 y3 = *reinterpret_cast<const f32x4*>(vs + 12);
        float yy[16] = { y0[0],y0[1],y0[2],y0[3], y1[0],y1[1],y1[2],y1[3],
                         y2[0],y2[1],y2[2],y2[3], y3[0],y3[1],y3[2],y3[3] };
        #pragma unroll
        for (int i = 0; i < 16; ++i)
            vt[(d_lin + i) * 32 + key_lin] = f2bf(yy[i]);
    }
    __syncthreads();
    {
        const int d  = tid >> 1;
        const int ko = (tid & 1) * 16;
        bf16x8 a = *reinterpret_cast<const bf16x8*>(&vt[d * 32 + ko + 0]);
        bf16x8 c = *reinterpret_cast<const bf16x8*>(&vt[d * 32 + ko + 8]);
        __bf16* vd = Vt + (size_t)b * D * NK + (size_t)d * NK + key0 + ko;
        *reinterpret_cast<bf16x8*>(vd + 0) = a;
        *reinterpret_cast<bf16x8*>(vd + 8) = c;
    }
}

// =====================================================================
// Main flash-attention kernel. 8 waves/block, 16 Q rows/wave.
// Async double-buffered K/V tiles; rowsum via ones-column WMMA;
// rowmax via v_permlane16 butterflies.
// =====================================================================
__global__ __launch_bounds__(256) void
fattn_kernel(const float* __restrict__ Qp, const __bf16* __restrict__ Kb,
             const __bf16* __restrict__ Vt, float* __restrict__ Op) {
    constexpr int D  = 128;
    constexpr int NQ = 2048;
    constexpr int NK = 2048;
    constexpr int T  = NK / 32;
    constexpr float SCALE = 0.08838834764831845f;   // 1/sqrt(128)

    __shared__ __align__(16) __bf16 ksh[2][32 * 128];   // [buf][key][d]
    __shared__ __align__(16) __bf16 vsh[2][128 * 32];   // [buf][d][key]
    __shared__ __align__(16) __bf16 psh[8 * 16 * 32];   // per-wave P scratch [m][key]

    const int tid  = threadIdx.x;
    const int lane = tid & 31;
    const int wave = tid >> 5;
    const int half = lane >> 4;
    const int l16  = lane & 15;
    const int b    = blockIdx.y;
    const int q_wg = blockIdx.x * 128;

    const char* kb_batch = (const char*)(Kb + (size_t)b * NK * D);
    const char* vt_batch = (const char*)(Vt + (size_t)b * D * NK);

    const unsigned klds0 = lds_off(&ksh[0][0]);
    const unsigned klds1 = lds_off(&ksh[1][0]);
    const unsigned vlds0 = lds_off(&vsh[0][0]);
    const unsigned vlds1 = lds_off(&vsh[1][0]);

    // 4 async b128 instructions per wave per tile (K contiguous; Vt row-strided)
    auto issue_tile = [&](int key0, unsigned klds, unsigned vlds) {
        const unsigned kgbase = (unsigned)key0 * (D * 2);
        #pragma unroll
        for (int i = 0; i < 2; ++i) {
            unsigned c = (unsigned)(wave * 64 + i * 32 + lane);
            async_b128(klds + c * 16, kb_batch, kgbase + c * 16);
        }
        #pragma unroll
        for (int i = 0; i < 2; ++i) {
            unsigned c = (unsigned)(wave * 64 + i * 32 + lane);
            unsigned d = c >> 2, part = c & 3;
            async_b128(vlds + c * 16, vt_batch, d * (NK * 2) + (unsigned)key0 * 2 + part * 16);
        }
    };

    issue_tile(0, klds0, vlds0);

    // ---- Q A-fragments, SCALE pre-folded ----
    bf16x16 qa[4];
    {
        const float* qrow = Qp + ((size_t)b * NQ + (q_wg + wave * 16 + l16)) * D;
        #pragma unroll
        for (int s = 0; s < 4; ++s) {
            bf16x8 g[2];
            #pragma unroll
            for (int gi = 0; gi < 2; ++gi) {
                const int d0 = s * 32 + half * 16 + gi * 8;
                f32x4 a = *reinterpret_cast<const f32x4*>(qrow + d0);
                f32x4 c = *reinterpret_cast<const f32x4*>(qrow + d0 + 4);
                bf16x8 t;
                t[0] = f2bf(a[0]*SCALE); t[1] = f2bf(a[1]*SCALE);
                t[2] = f2bf(a[2]*SCALE); t[3] = f2bf(a[3]*SCALE);
                t[4] = f2bf(c[0]*SCALE); t[5] = f2bf(c[1]*SCALE);
                t[6] = f2bf(c[2]*SCALE); t[7] = f2bf(c[3]*SCALE);
                g[gi] = t;
            }
            qa[s] = CAT16(g[0], g[1]);
        }
    }

    // constant all-ones B fragment (rowsum-by-WMMA), lives in registers
    bf16x16 ones;
    #pragma unroll
    for (int i = 0; i < 16; ++i) ones[i] = f2bf(1.0f);

    f32x8 oacc[8];
    #pragma unroll
    for (int i = 0; i < 8; ++i) oacc[i] = (f32x8)(0.0f);
    f32x8 lacc = (f32x8)(0.0f);        // running softmax denominator (broadcast per row)
    float m_run[8];
    #pragma unroll
    for (int r = 0; r < 8; ++r) m_run[r] = -__builtin_inff();

    __bf16* pbase = psh + wave * (16 * 32);

    for (int t = 0; t < T; ++t) {
        const int buf = t & 1;
        const __bf16* kcur = &ksh[buf][0];
        const __bf16* vcur = &vsh[buf][0];

        if (t + 1 < T) {
            issue_tile((t + 1) * 32, (buf ? klds0 : klds1), (buf ? vlds0 : vlds1));
            asm volatile("s_wait_asynccnt 4" ::: "memory");
        } else {
            asm volatile("s_wait_asynccnt 0" ::: "memory");
        }
        __syncthreads();

        // ---- load ALL K fragments first, then the 8-WMMA chain ----
        bf16x16 kf[2][4];
        #pragma unroll
        for (int n = 0; n < 2; ++n) {
            const int keyc = (n * 16 + l16) * 128;
            #pragma unroll
            for (int s = 0; s < 4; ++s) {
                bf16x8 g0 = *reinterpret_cast<const bf16x8*>(&kcur[keyc + s * 32 + half * 16 + 0]);
                bf16x8 g1 = *reinterpret_cast<const bf16x8*>(&kcur[keyc + s * 32 + half * 16 + 8]);
                kf[n][s] = CAT16(g0, g1);
            }
        }
        f32x8 sc[2];
        #pragma unroll
        for (int n = 0; n < 2; ++n) {
            f32x8 c = (f32x8)(0.0f);
            #pragma unroll
            for (int s = 0; s < 4; ++s)
                c = __builtin_amdgcn_wmma_f32_16x16x32_bf16(
                        false, qa[s], false, kf[n][s], (short)0, c, false, false);
            sc[n] = c;
        }

        // ---- online softmax: rowmax via permlane16 butterflies (VALU only) ----
        float mnew[8], alpha[8];
        #pragma unroll
        for (int r = 0; r < 8; ++r) {
            float v = fmaxf(sc[0][r], sc[1][r]);
            v = xmax16(v, 0x67452301u, 0xEFCDAB89u, 1);
            v = xmax16(v, 0x54761032u, 0xDCFE98BAu, 2);
            v = xmax16(v, 0x32107654u, 0xBA98FEDCu, 4);
            v = xmax16(v, 0xFEDCBA98u, 0x76543210u, 8);
            mnew[r]  = fmaxf(m_run[r], v);
            alpha[r] = __expf(m_run[r] - mnew[r]);
            m_run[r] = mnew[r];
        }
        #pragma unroll
        for (int n = 0; n < 2; ++n)
            #pragma unroll
            for (int r = 0; r < 8; ++r) sc[n][r] = __expf(sc[n][r] - mnew[r]);

        // rescale accumulators (lacc rides along; rowsum added by the ones-WMMA)
        #pragma unroll
        for (int r = 0; r < 8; ++r) {
            lacc[r] *= alpha[r];
            #pragma unroll
            for (int nt = 0; nt < 8; ++nt) oacc[nt][r] *= alpha[r];
        }

        // ---- transpose P (D layout -> A layout) via per-wave LDS scratch ----
        #pragma unroll
        for (int n = 0; n < 2; ++n)
            #pragma unroll
            for (int r = 0; r < 8; ++r)
                pbase[(r + half * 8) * 32 + n * 16 + l16] = f2bf(sc[n][r]);
        asm volatile("s_wait_dscnt 0" ::: "memory");
        bf16x8 p0 = *reinterpret_cast<const bf16x8*>(&pbase[l16 * 32 + half * 16]);
        bf16x8 p1 = *reinterpret_cast<const bf16x8*>(&pbase[l16 * 32 + half * 16 + 8]);
        bf16x16 pa = CAT16(p0, p1);

        // ---- load ALL V fragments, then PV chain + rowsum WMMA ----
        bf16x16 vf[8];
        #pragma unroll
        for (int nt = 0; nt < 8; ++nt) {
            const int dc = (nt * 16 + l16) * 32;
            bf16x8 v0 = *reinterpret_cast<const bf16x8*>(&vcur[dc + half * 16 + 0]);
            bf16x8 v1 = *reinterpret_cast<const bf16x8*>(&vcur[dc + half * 16 + 8]);
            vf[nt] = CAT16(v0, v1);
        }
        #pragma unroll
        for (int nt = 0; nt < 8; ++nt)
            oacc[nt] = __builtin_amdgcn_wmma_f32_16x16x32_bf16(
                           false, pa, false, vf[nt], (short)0, oacc[nt], false, false);
        lacc = __builtin_amdgcn_wmma_f32_16x16x32_bf16(
                   false, pa, false, ones, (short)0, lacc, false, false);

        __syncthreads();
    }

    // ---- epilogue: O = acc / l  (lacc already broadcast across lanes per row) ----
    #pragma unroll
    for (int r = 0; r < 8; ++r) {
        const float inv = 1.0f / lacc[r];
        const int m = r + half * 8;
        const size_t row = ((size_t)b * NQ + q_wg + wave * 16 + m) * D;
        #pragma unroll
        for (int nt = 0; nt < 8; ++nt)
            Op[row + nt * 16 + l16] = oacc[nt][r] * inv;
    }
}

extern "C" void kernel_launch(void* const* d_in, const int* in_sizes, int n_in,
                              void* d_out, int out_size, void* d_ws, size_t ws_size,
                              hipStream_t stream) {
    const float* q = (const float*)d_in[0];
    const float* k = (const float*)d_in[1];
    const float* v = (const float*)d_in[2];
    float* out = (float*)d_out;

    const int B = in_sizes[0] / (2048 * 128);   // 16
    __bf16* Kb = (__bf16*)d_ws;
    __bf16* Vt = (__bf16*)((char*)d_ws + (size_t)B * 2048 * 128 * 2);

    dim3 pgrid(2048 / 32, B), pblock(256);
    prep_kernel<<<pgrid, pblock, 0, stream>>>(k, v, Kb, Vt);

    dim3 grid(2048 / 128, B), block(256);
    fattn_kernel<<<grid, block, 0, stream>>>(q, Kb, Vt, out);
}
